// CorrectMaskedEfficientViTBlock_3238405341443
// MI455X (gfx1250) — matrix operand order repaired and checked by reading, not compiled, not run on hardware
//
#include <hip/hip_runtime.h>
#include <stdint.h>
#include <stddef.h>

// ---------------------------------------------------------------------------
// MaskedEfficientViT block for MI455X (gfx1250, wave32, WMMA).
// bf16 WMMA (16x16x32, f32 accum) GEMMs; token-drop factorization (GEMM only
// the 1024 kept columns where possible; masked rows of the projection are a
// precomputed constant vector). 128x64 block tile, 32x32 wave tile.
// Async global->LDS staging (ASYNCcnt) via global_load_async_to_lds_b128.
// ---------------------------------------------------------------------------

#define Bn   8
#define Cn   256
#define Hn   64
#define Wn   64
#define Ln   4096      // H*W
#define KEEP 1024
#define HEADSn 32
#define EXPn 1024

typedef __attribute__((ext_vector_type(16))) __bf16          bf16x16;
typedef __attribute__((ext_vector_type(8)))  float           f32x8;
typedef __attribute__((ext_vector_type(16))) unsigned short  u16x16;
typedef int g4i __attribute__((__vector_size__(16)));   // matches builtin pointee

#if defined(__has_builtin)
#if __has_builtin(__builtin_amdgcn_global_load_async_to_lds_b128)
#define HAVE_ASYNC_LDS 1
#endif
#endif
#ifndef HAVE_ASYNC_LDS
#define HAVE_ASYNC_LDS 0
#endif

__device__ __forceinline__ unsigned short f2bf(float f) {
  union { float f; unsigned u; } c; c.f = f;
  unsigned u = c.u;
  u += 0x7fffu + ((u >> 16) & 1u);           // round-to-nearest-even
  return (unsigned short)(u >> 16);
}
__device__ __forceinline__ float bf2f(unsigned short h) {
  union { unsigned u; float f; } c; c.u = ((unsigned)h) << 16;
  return c.f;
}
__device__ __forceinline__ float hswish(float x) {
  float t = fminf(fmaxf(x + 3.0f, 0.0f), 6.0f);
  return x * t * (1.0f / 6.0f);
}

// 16-byte global->LDS stage: async DMA when available, vector copy otherwise.
__device__ __forceinline__ void stage16(unsigned short* lds, const unsigned short* g) {
#if HAVE_ASYNC_LDS
  __builtin_amdgcn_global_load_async_to_lds_b128(
      (__attribute__((address_space(1))) g4i*)(uintptr_t)g,
      (__attribute__((address_space(3))) g4i*)(unsigned)(uintptr_t)lds,
      0, 0);
#else
  *(uint4*)lds = *(const uint4*)g;
#endif
}
__device__ __forceinline__ void stage_fence() {
#if HAVE_ASYNC_LDS
#if __has_builtin(__builtin_amdgcn_s_wait_asynccnt)
  __builtin_amdgcn_s_wait_asynccnt(0);
#else
  asm volatile("s_wait_asynccnt 0" ::: "memory");
#endif
#endif
}

// ---------------------------------------------------------------------------
// 1) Bitonic argsort of noise per batch (4096 keys in LDS).
// ---------------------------------------------------------------------------
__global__ __launch_bounds__(1024) void k_sort(const float* __restrict__ noise,
                                               int* __restrict__ ids_keep,
                                               int* __restrict__ slot_of_pos) {
  __shared__ float key[Ln];
  __shared__ int   idx[Ln];
  const int b = blockIdx.x, tid = threadIdx.x;
  for (int i = tid; i < Ln; i += 1024) { key[i] = noise[b * Ln + i]; idx[i] = i; }
  __syncthreads();
  for (int k = 2; k <= Ln; k <<= 1) {
    for (int j = k >> 1; j > 0; j >>= 1) {
      for (int i = tid; i < Ln; i += 1024) {
        int ixj = i ^ j;
        if (ixj > i) {
          bool up = ((i & k) == 0);
          float a = key[i], c = key[ixj];
          bool sw = up ? (a > c) : (a < c);
          if (sw) {
            key[i] = c; key[ixj] = a;
            int t = idx[i]; idx[i] = idx[ixj]; idx[ixj] = t;
          }
        }
      }
      __syncthreads();
    }
  }
  if (tid < KEEP) ids_keep[b * KEEP + tid] = idx[tid];
  for (int i = tid; i < Ln; i += 1024)
    slot_of_pos[b * Ln + idx[i]] = (i < KEEP) ? i : -1;
}

// ---------------------------------------------------------------------------
// 2) f32 -> bf16 conversion (weights) and mask-token projection constant
// ---------------------------------------------------------------------------
__global__ void k_convert(const float* __restrict__ in, unsigned short* __restrict__ out, int n) {
  int i = blockIdx.x * 256 + threadIdx.x;
  if (i < n) out[i] = f2bf(in[i]);
}

__global__ void k_maskproj(const float* __restrict__ Wproj, const float* __restrict__ mtok,
                           float* __restrict__ mask_proj) {
  int o = threadIdx.x;
  float s = 0.f;
  for (int c = 0; c < Cn; ++c) s += Wproj[o * Cn + c] * mtok[c];
  mask_proj[o] = s;
}

// ---------------------------------------------------------------------------
// 3) Gather kept columns of x, K-fast layout: x_vis[b][j][c] (bf16).
//    Gather reads are random in pos anyway; c-fast writes are coalesced.
// ---------------------------------------------------------------------------
__global__ void k_gather(const float* __restrict__ x, const int* __restrict__ ids_keep,
                         unsigned short* __restrict__ xvis) {
  int i = blockIdx.x * 256 + threadIdx.x;        // i < B*KEEP*C = 2^21
  int c = i & (Cn - 1);
  int j = (i >> 8) & (KEEP - 1);
  int b = i >> 18;
  int pos = ids_keep[b * KEEP + j];
  xvis[i] = f2bf(x[((size_t)b * Cn + c) * Ln + pos]);
}

// ---------------------------------------------------------------------------
// 4) bf16 WMMA GEMM:  Out[b][M][N] = A[M,K] x B[b]
//    BTRANS=0: global B is [N,K] (K-fast)  -> direct (async) staging
//    BTRANS=1: global B is [K,N] (N-fast)  -> transpose scatter staging
//    Block tile 128x64, K-step 32, 8 waves in a 4x2 grid, 32x32 per wave
//    (4 x v_wmma_f32_16x16x32_bf16 per wave per K-step; A/B frags reused 2x).
//    EPI 0: f32 | EPI 1: hswish*inv -> bf16 | EPI 2: (+resid)*inv -> f32
// ---------------------------------------------------------------------------
template <int EPI, int BTRANS>
__global__ __launch_bounds__(256) void k_gemm(
    const unsigned short* __restrict__ A,
    const unsigned short* __restrict__ Bm,
    int M, int K, int N,
    float* __restrict__ OutF,
    unsigned short* __restrict__ OutB,
    const uint8_t* __restrict__ smask,
    const float* __restrict__ Resid) {
  __shared__ __align__(16) unsigned short sA[128][32];  // [m][k]
  __shared__ __align__(16) unsigned short sB[64][32];   // [n][k]
  const int b = blockIdx.z;
  const int m0 = blockIdx.y * 128, n0 = blockIdx.x * 64;
  const int tid = threadIdx.x;
  const int wave = tid >> 5, lane = tid & 31;
  const int wm = wave >> 1, wn = wave & 1;
  const int half = lane >> 4, l16 = lane & 15;
  const unsigned short* Bb = Bm + (size_t)b * K * N;

  const int ar = tid >> 2,  acs = (tid & 3) << 3;   // A: rows ar, ar+64; 4 segs of 8
  const int btr = tid >> 3, bts = (tid & 7) << 3;   // BTRANS=1: 32 k-rows x 8 segs
  const int bnr = tid >> 2, bks = (tid & 3) << 3;   // BTRANS=0: 64 n-rows x 4 segs

  f32x8 acc00 = {}, acc01 = {}, acc10 = {}, acc11 = {};
  for (int k0 = 0; k0 < K; k0 += 32) {
    // --- stage A (2 x 16B per thread) ---
    stage16(&sA[ar][acs],      A + (size_t)(m0 + ar) * K + k0 + acs);
    stage16(&sA[64 + ar][acs], A + (size_t)(m0 + 64 + ar) * K + k0 + acs);
    // --- stage B ---
    if (BTRANS == 0) {
      stage16(&sB[bnr][bks], Bb + (size_t)(n0 + bnr) * K + k0 + bks);
    } else {
      union { uint4 v4; unsigned short s[8]; } t;
      t.v4 = *(const uint4*)(Bb + (size_t)(k0 + btr) * N + n0 + bts);
#pragma unroll
      for (int i = 0; i < 8; ++i) sB[bts + i][btr] = t.s[i];
    }
    if (k0 + 32 < K) {   // next K tile -> global_prefetch_b8
      __builtin_prefetch(A + (size_t)(m0 + ar) * K + k0 + 32 + acs, 0, 0);
      if (BTRANS == 0)
        __builtin_prefetch(Bb + (size_t)(n0 + bnr) * K + k0 + 32 + bks, 0, 0);
      else
        __builtin_prefetch(Bb + (size_t)(k0 + 32 + btr) * N + n0 + bts, 0, 0);
    }
    stage_fence();
    __syncthreads();

    // --- fragments per ISA VGPR layout tables ---
    // A 16x32: lane=m+16h; elems 0..7 -> k=8h+e, elems 8..15 -> k=16+8h+e
    // B 32x16: lane=n+16h; elems 0..7 -> k=8h+e, elems 8..15 -> k=16+8h+e
    u16x16 a0u, a1u, b0u, b1u;
    const unsigned short* pa0 = &sA[wm * 32 + l16][0];
    const unsigned short* pa1 = &sA[wm * 32 + 16 + l16][0];
    const unsigned short* pb0 = &sB[wn * 32 + l16][0];
    const unsigned short* pb1 = &sB[wn * 32 + 16 + l16][0];
#pragma unroll
    for (int e = 0; e < 8; ++e) {
      a0u[e]     = pa0[8 * half + e];
      a0u[8 + e] = pa0[16 + 8 * half + e];
      a1u[e]     = pa1[8 * half + e];
      a1u[8 + e] = pa1[16 + 8 * half + e];
    }
#pragma unroll
    for (int e = 0; e < 8; ++e) {
      b0u[e]     = pb0[8 * half + e];
      b0u[8 + e] = pb0[16 + 8 * half + e];
      b1u[e]     = pb1[8 * half + e];
      b1u[8 + e] = pb1[16 + 8 * half + e];
    }
    bf16x16 a0 = __builtin_bit_cast(bf16x16, a0u);
    bf16x16 a1 = __builtin_bit_cast(bf16x16, a1u);
    bf16x16 b0 = __builtin_bit_cast(bf16x16, b0u);
    bf16x16 b1 = __builtin_bit_cast(bf16x16, b1u);
    acc00 = __builtin_amdgcn_wmma_f32_16x16x32_bf16(false, a0, false, b0, (short)0, acc00, false, false);
    acc01 = __builtin_amdgcn_wmma_f32_16x16x32_bf16(false, a0, false, b1, (short)0, acc01, false, false);
    acc10 = __builtin_amdgcn_wmma_f32_16x16x32_bf16(false, a1, false, b0, (short)0, acc10, false, false);
    acc11 = __builtin_amdgcn_wmma_f32_16x16x32_bf16(false, a1, false, b1, (short)0, acc11, false, false);
    __syncthreads();
  }

  // --- epilogue: C/D layout: vgpr r, lanes<16 -> M=r, lanes>=16 -> M=r+8; N=l16
#pragma unroll
  for (int r = 0; r < 8; ++r) {
    int ma = m0 + wm * 32 + r + 8 * half;
    int mb = ma + 16;
    int na = n0 + wn * 32 + l16;
    int nb = na + 16;
    float v[4] = { acc00[r], acc01[r], acc10[r], acc11[r] };
    int   mm[4] = { ma, ma, mb, mb };
    int   nn[4] = { na, nb, na, nb };
#pragma unroll
    for (int q = 0; q < 4; ++q) {
      size_t o = (size_t)b * M * N + (size_t)mm[q] * N + nn[q];
      if (EPI == 0) {
        OutF[o] = v[q];
      } else if (EPI == 1) {
        float iv = smask[(size_t)b * N + nn[q]] ? 0.f : 1.f;
        OutB[o] = f2bf(hswish(v[q]) * iv);
      } else {
        OutF[o] = smask[(size_t)b * N + nn[q]] ? 0.f : (v[q] + Resid[o]);
      }
    }
  }
}

// ---------------------------------------------------------------------------
// 5) Linear attention per (b, head): kv[9][8] = v1 k^T over 1024 tokens, then
//    attn = (kv[:8] q) / (kv[8] q + eps). Output K-fast: attn_vis[b][n][c],
//    one packed 16B store per token per thread.
// ---------------------------------------------------------------------------
__global__ __launch_bounds__(256) void k_attn(const float* __restrict__ qkv,
                                              unsigned short* __restrict__ attnb) {
  extern __shared__ float s[];                    // [24][1024]
  __shared__ float skv[9][8];
  const int b = blockIdx.x >> 5, h = blockIdx.x & 31;
  const float* base = qkv + ((size_t)b * 768 + h * 24) * KEEP;
  for (int e = threadIdx.x; e < 24 * KEEP; e += 256) {
    int t = e >> 10;
    float v = base[e];
    if (t < 16) v = fmaxf(v, 0.0f);               // relu on q and k
    s[e] = v;
  }
  __syncthreads();
  if (threadIdx.x < 72) {
    int d = threadIdx.x >> 3, e2 = threadIdx.x & 7;
    const float* krow = s + (8 + e2) * KEEP;
    float acc = 0.f;
    if (d == 8) {
      for (int n = 0; n < KEEP; ++n) acc += krow[n];
    } else {
      const float* vrow = s + (16 + d) * KEEP;
      for (int n = 0; n < KEEP; ++n) acc += vrow[n] * krow[n];
    }
    skv[d][e2] = acc;
  }
  __syncthreads();
  for (int n = threadIdx.x; n < KEEP; n += 256) {
    float q[8];
#pragma unroll
    for (int e = 0; e < 8; ++e) q[e] = s[e * KEEP + n];
    float den = 0.f;
#pragma unroll
    for (int e = 0; e < 8; ++e) den += skv[8][e] * q[e];
    float rden = 1.0f / (den + 1e-15f);
    union { uint4 v4; unsigned short u[8]; } pk;
#pragma unroll
    for (int d = 0; d < 8; ++d) {
      float num = 0.f;
#pragma unroll
      for (int e = 0; e < 8; ++e) num += skv[d][e] * q[e];
      pk.u[d] = f2bf(num * rden);
    }
    *(uint4*)(attnb + ((size_t)b * KEEP + n) * Cn + h * 8) = pk.v4;  // 16B aligned
  }
}

// ---------------------------------------------------------------------------
// 6) Scatter projection back to full grid + residual; masked x_ctx stored
//    f32 [c][pos] (residual path) and bf16 [c][pos] (GEMM3 B, BTRANS=1).
// ---------------------------------------------------------------------------
__global__ void k_scatter(const float* __restrict__ proj_vis, const float* __restrict__ mask_proj,
                          const float* __restrict__ x, const uint8_t* __restrict__ smask,
                          const int* __restrict__ slot_of_pos,
                          float* __restrict__ xctx_f, unsigned short* __restrict__ xctx_b) {
  int i = blockIdx.x * 256 + threadIdx.x;         // i < B*C*L = 2^23
  int pos = i & (Ln - 1);
  int c = (i >> 12) & (Cn - 1);
  int b = i >> 20;
  int slot = slot_of_pos[b * Ln + pos];
  float v = (slot >= 0) ? proj_vis[((size_t)b * Cn + c) * KEEP + slot] : mask_proj[c];
  float invf = smask[b * Ln + pos] ? 0.f : 1.f;
  float xc = (v + x[i]) * invf;
  xctx_f[i] = xc;
  xctx_b[i] = f2bf(xc);
}

// ---------------------------------------------------------------------------
// 7) out_mask[b][pos] = 1 iff no masked pixel in the 3x3 neighborhood
// ---------------------------------------------------------------------------
__global__ void k_outmask(const uint8_t* __restrict__ smask, uint8_t* __restrict__ omask) {
  int i = blockIdx.x * 256 + threadIdx.x;         // i < B*L
  int pos = i & (Ln - 1), b = i >> 12;
  int y = pos >> 6, x = pos & 63;
  int any = 0;
  for (int dy = -1; dy <= 1; ++dy)
    for (int dx = -1; dx <= 1; ++dx) {
      int yy = y + dy, xx = x + dx;
      if (yy >= 0 && yy < Hn && xx >= 0 && xx < Wn)
        any |= smask[b * Ln + yy * Wn + xx];
    }
  omask[i] = any ? 0 : 1;
}

// ---------------------------------------------------------------------------
// 8) Depthwise 3x3 over bf16 x1 (67MB -> resident in 192MB L2, so the 9x
//    re-reads stay on-chip), x2 = hswish(conv) * out_mask, bf16 [e][pos].
// ---------------------------------------------------------------------------
__global__ void k_dw(const unsigned short* __restrict__ x1, const float* __restrict__ Wdw,
                     const uint8_t* __restrict__ omask, unsigned short* __restrict__ x2) {
  int i = blockIdx.x * 256 + threadIdx.x;         // i < B*EXP*L = 2^25
  int pos = i & (Ln - 1);
  int ch = (i >> 12) & (EXPn - 1);
  int b = i >> 22;
  if (!omask[b * Ln + pos]) { x2[i] = 0; return; }
  int y = pos >> 6, x = pos & 63;
  const unsigned short* plane = x1 + ((size_t)b * EXPn + ch) * Ln;
  const float* w = Wdw + ch * 9;
  float acc = 0.f;
#pragma unroll
  for (int dy = -1; dy <= 1; ++dy) {
    int yy = y + dy;
    if (yy < 0 || yy >= Hn) continue;
#pragma unroll
    for (int dx = -1; dx <= 1; ++dx) {
      int xx = x + dx;
      if (xx < 0 || xx >= Wn) continue;
      acc += w[(dy + 1) * 3 + (dx + 1)] * bf2f(plane[yy * Wn + xx]);
    }
  }
  x2[i] = f2bf(hswish(acc));
}

// ---------------------------------------------------------------------------
// Launch
// ---------------------------------------------------------------------------
static inline int cdiv(int a, int b) { return (a + b - 1) / b; }

extern "C" void kernel_launch(void* const* d_in, const int* in_sizes, int n_in,
                              void* d_out, int out_size, void* d_ws, size_t ws_size,
                              hipStream_t stream) {
  const float*   x     = (const float*)d_in[0];
  const uint8_t* smask = (const uint8_t*)d_in[1];   // jnp bool_: 1 byte/elem
  const float*   noise = (const float*)d_in[2];
  const float*   Wqkv  = (const float*)d_in[3];
  const float*   Wproj = (const float*)d_in[4];
  const float*   mtok  = (const float*)d_in[5];
  const float*   Winv  = (const float*)d_in[6];
  const float*   Wdw   = (const float*)d_in[7];
  const float*   Wpw   = (const float*)d_in[8];
  float* out = (float*)d_out;

  // ---- carve workspace (~228 MB) ----
  char* p = (char*)d_ws;
  auto carve = [&](size_t bytes) -> char* {
    char* r = p; p += (bytes + 255) & ~(size_t)255; return r;
  };
  int*            ids_keep   = (int*)carve((size_t)Bn * KEEP * 4);
  int*            slot_pos   = (int*)carve((size_t)Bn * Ln * 4);
  unsigned short* Wqkv_b     = (unsigned short*)carve((size_t)768 * Cn * 2);
  unsigned short* Wproj_b    = (unsigned short*)carve((size_t)Cn * Cn * 2);
  unsigned short* Winv_b     = (unsigned short*)carve((size_t)EXPn * Cn * 2);
  unsigned short* Wpw_b      = (unsigned short*)carve((size_t)Cn * EXPn * 2);
  float*          mask_proj  = (float*)carve((size_t)Cn * 4);
  unsigned short* x_vis      = (unsigned short*)carve((size_t)Bn * KEEP * Cn * 2); // [b][j][c]
  float*          qkv_vis    = (float*)carve((size_t)Bn * 768 * KEEP * 4);         // [b][o][j]
  unsigned short* attn_vis   = (unsigned short*)carve((size_t)Bn * KEEP * Cn * 2); // [b][n][c]
  float*          proj_vis   = (float*)carve((size_t)Bn * Cn * KEEP * 4);          // [b][c][j]
  float*          xctx_f     = (float*)carve((size_t)Bn * Cn * Ln * 4);            // [b][c][pos]
  unsigned short* xctx_b     = (unsigned short*)carve((size_t)Bn * Cn * Ln * 2);   // [b][c][pos]
  unsigned short* x1         = (unsigned short*)carve((size_t)Bn * EXPn * Ln * 2); // [b][e][pos]
  uint8_t*        omask      = (uint8_t*)carve((size_t)Bn * Ln);
  unsigned short* x2         = (unsigned short*)carve((size_t)Bn * EXPn * Ln * 2); // [b][e][pos]
  (void)in_sizes; (void)n_in; (void)out_size; (void)ws_size;

  // weight conversions
  k_convert<<<cdiv(768 * Cn, 256), 256, 0, stream>>>(Wqkv, Wqkv_b, 768 * Cn);
  k_convert<<<cdiv(Cn * Cn, 256), 256, 0, stream>>>(Wproj, Wproj_b, Cn * Cn);
  k_convert<<<cdiv(EXPn * Cn, 256), 256, 0, stream>>>(Winv, Winv_b, EXPn * Cn);
  k_convert<<<cdiv(Cn * EXPn, 256), 256, 0, stream>>>(Wpw, Wpw_b, Cn * EXPn);
  k_maskproj<<<1, 256, 0, stream>>>(Wproj, mtok, mask_proj);

  // token shuffle
  k_sort<<<Bn, 1024, 0, stream>>>(noise, ids_keep, slot_pos);
  k_gather<<<cdiv(Bn * KEEP * Cn, 256), 256, 0, stream>>>(x, ids_keep, x_vis);

  // qkv on kept tokens: [768,256] x [256,1024], B is [N,K] -> async staging
  k_gemm<0, 0><<<dim3(KEEP / 64, 768 / 128, Bn), 256, 0, stream>>>(
      Wqkv_b, x_vis, 768, Cn, KEEP, qkv_vis, nullptr, nullptr, nullptr);

  // linear attention (24 rows * 1024 * 4B dynamic LDS)
  k_attn<<<Bn * HEADSn, 256, 24 * KEEP * 4, stream>>>(qkv_vis, attn_vis);

  // projection on kept tokens: [256,256] x [256,1024], B is [N,K]
  k_gemm<0, 0><<<dim3(KEEP / 64, Cn / 128, Bn), 256, 0, stream>>>(
      Wproj_b, attn_vis, Cn, Cn, KEEP, proj_vis, nullptr, nullptr, nullptr);

  // scatter + residual -> x_ctx (masked)
  k_scatter<<<cdiv(Bn * Cn * Ln, 256), 256, 0, stream>>>(
      proj_vis, mask_proj, x, smask, slot_pos, xctx_f, xctx_b);

  // expand: x1 = hswish([1024,256] x [256,4096]) * inv  (bf16 out), B is [K,N]
  k_gemm<1, 1><<<dim3(Ln / 64, EXPn / 128, Bn), 256, 0, stream>>>(
      Winv_b, xctx_b, EXPn, Cn, Ln, nullptr, x1, smask, nullptr);

  // depthwise 3x3 + hswish * out_mask
  k_outmask<<<cdiv(Bn * Ln, 256), 256, 0, stream>>>(smask, omask);
  k_dw<<<cdiv(Bn * EXPn * Ln, 256), 256, 0, stream>>>(x1, Wdw, omask, x2);

  // project: out = ([256,1024] x [1024,4096] + x_ctx) * inv  (f32 out), B is [K,N]
  k_gemm<2, 1><<<dim3(Ln / 64, Cn / 128, Bn), 256, 0, stream>>>(
      Wpw_b, x2, Cn, EXPn, Ln, out, nullptr, smask, xctx_f);
}
